// TransformerNet_1279900254271
// MI455X (gfx1250) — matrix operand order repaired and checked by reading, not compiled, not run on hardware
//
#include <hip/hip_runtime.h>
#include <hip/hip_bf16.h>
#include <math.h>

typedef __attribute__((ext_vector_type(16))) _Float16 v16h;
typedef __attribute__((ext_vector_type(8)))  float    v8f;

union FragU { v16h v; _Float16 e[16]; };

// scheduler pinning: mask 0x20 = VMEM read, 0x8 = MFMA/WMMA
#if defined(__has_builtin)
#  if __has_builtin(__builtin_amdgcn_sched_group_barrier)
#    define SGB(mask, size, sync) __builtin_amdgcn_sched_group_barrier(mask, size, sync)
#  else
#    define SGB(mask, size, sync)
#  endif
#else
#  define SGB(mask, size, sync)
#endif

// ---------------- helpers ----------------
static __device__ __forceinline__ unsigned int enc_f(float f){
    unsigned int u = __float_as_uint(f);
    return (u & 0x80000000u) ? ~u : (u | 0x80000000u);   // order-preserving float->uint
}
static __device__ __forceinline__ float dec_f(unsigned int u){
    return (u & 0x80000000u) ? __uint_as_float(u ^ 0x80000000u) : __uint_as_float(~u);
}
static inline int cdiv(long long a, long long b){ return (int)((a + b - 1) / b); }

// A fragment (16x32 f16): lane<16 -> row tn*16+lane, K {0..15}; lane>=16 -> K {16..31}
static __device__ __forceinline__ void load_fragA(FragU& u, const _Float16* __restrict__ p){
    #pragma unroll
    for (int i = 0; i < 16; ++i){ u.e[i] = p[i]; }
}
// B fragment (32x16 f16, from col-major/transposed weight): 16 contiguous K per lane
static __device__ __forceinline__ void load_fragB(FragU& u, const _Float16* __restrict__ p){
    #pragma unroll
    for (int i = 0; i < 16; ++i){ u.e[i] = p[i]; }
}
static __device__ __forceinline__ v8f wmma_f16(const FragU& a, const FragU& b, v8f c){
    return __builtin_amdgcn_wmma_f32_16x16x32_f16(false, a.v, false, b.v,
                                                  (short)0, c, false, false);
}

// ---------------- elementwise / fill ----------------
__global__ void k_f32_to_f16(const float* __restrict__ s, _Float16* __restrict__ d, int n){
    int i = blockIdx.x * blockDim.x + threadIdx.x;
    if (i < n) d[i] = (_Float16)s[i];
}
__global__ void k_transpose_f16(const float* __restrict__ W, _Float16* __restrict__ Wt, int K, int M){
    int i = blockIdx.x * blockDim.x + threadIdx.x;
    if (i < K * M){ int k = i / M, m = i - k * M; Wt[(size_t)m * K + k] = (_Float16)W[i]; }
}
__global__ void k_fill_f32(float* p, float v, int n){
    int i = blockIdx.x * blockDim.x + threadIdx.x;
    if (i < n) p[i] = v;
}
__global__ void k_fill_u32(unsigned int* p, unsigned int v, int n){
    int i = blockIdx.x * blockDim.x + threadIdx.x;
    if (i < n) p[i] = v;
}
__global__ void k_elu(float* x, int n){
    int i = blockIdx.x * blockDim.x + threadIdx.x;
    if (i < n){ float v = x[i]; x[i] = v > 0.f ? v : (expf(v) - 1.f); }
}

// ---------------- fused WMMA GEMM for the 4 projections ----------------
// Out[m][N,M] = A[N,K](f16) @ Bt[m][M,K](f16)^T + bias[m][M], m = q,k,v,s
// One wave: 16-row x 32-col tile for ALL FOUR weight matrices (8 accumulators).
// sched_group_barrier pins the schedule so fragment loads run >=1 group ahead
// of the consuming WMMAs -> distinct VGPRs, partial s_wait_loadcnt, and
// WMMA execution overlapped with in-flight loads.
__global__ void k_gemm_qkvs(const _Float16* __restrict__ A,
                            const _Float16* __restrict__ BtQ, const _Float16* __restrict__ BtK,
                            const _Float16* __restrict__ BtV, const _Float16* __restrict__ BtS,
                            const float* __restrict__ bQ, const float* __restrict__ bK,
                            const float* __restrict__ bV, const float* __restrict__ bS,
                            float* __restrict__ OQ, float* __restrict__ OK,
                            float* __restrict__ OV, float* __restrict__ OS,
                            int Nrows, int Kdim, int M){
    int wid  = (blockIdx.x * blockDim.x + threadIdx.x) >> 5;
    int lane = threadIdx.x & 31;
    const int tilesM2 = M >> 5;            // 32-col tiles
    int tn  = wid / tilesM2;               // row tile (nodes)
    int tm2 = wid - tn * tilesM2;          // 32-col tile
    if (tn * 16 >= Nrows) return;

    const bool hi  = lane >= 16;
    const int  l15 = lane & 15;
    const int  arow = tn * 16 + l15;
    const int  col0 = tm2 * 32 + l15;      // first 16-col tile's column for this lane
    const int  col1 = col0 + 16;           // second 16-col tile's column

    const size_t bro = (size_t)(hi ? 16 : 0);
    const _Float16* ap  = A   + (size_t)arow * Kdim + (hi ? 16 : 0);
    const _Float16* q0  = BtQ + (size_t)col0 * Kdim + bro;
    const _Float16* q1  = BtQ + (size_t)col1 * Kdim + bro;
    const _Float16* k0p = BtK + (size_t)col0 * Kdim + bro;
    const _Float16* k1p = BtK + (size_t)col1 * Kdim + bro;
    const _Float16* v0  = BtV + (size_t)col0 * Kdim + bro;
    const _Float16* v1  = BtV + (size_t)col1 * Kdim + bro;
    const _Float16* s0  = BtS + (size_t)col0 * Kdim + bro;
    const _Float16* s1  = BtS + (size_t)col1 * Kdim + bro;

    v8f aq0 = {}, aq1 = {}, ak0 = {}, ak1 = {}, av0 = {}, av1 = {}, as0 = {}, as1 = {};

    for (int kk = 0; kk < Kdim; kk += 32){
        FragU a, bq0f, bq1f, bk0f, bk1f, bv0f, bv1f, bs0f, bs1f;
        load_fragA(a, ap + kk);
        load_fragB(bq0f, q0  + kk);  load_fragB(bq1f, q1  + kk);
        load_fragB(bk0f, k0p + kk);  load_fragB(bk1f, k1p + kk);
        load_fragB(bv0f, v0 + kk);   load_fragB(bv1f, v1 + kk);
        load_fragB(bs0f, s0 + kk);   load_fragB(bs1f, s1 + kk);
        aq0 = wmma_f16(a, bq0f, aq0);
        aq1 = wmma_f16(a, bq1f, aq1);
        ak0 = wmma_f16(a, bk0f, ak0);
        ak1 = wmma_f16(a, bk1f, ak1);
        av0 = wmma_f16(a, bv0f, av0);
        av1 = wmma_f16(a, bv1f, av1);
        as0 = wmma_f16(a, bs0f, as0);
        as1 = wmma_f16(a, bs1f, as1);
        // pinned schedule: loads stay >=1 stage ahead of their WMMAs
        SGB(0x020, 10, 0);   // A(2) + q-pair(4) + k-pair(4) vmem reads
        SGB(0x008,  2, 0);   // q WMMAs (overlap v loads)
        SGB(0x020,  4, 0);   // v-pair loads
        SGB(0x008,  2, 0);   // k WMMAs (overlap s loads)
        SGB(0x020,  4, 0);   // s-pair loads
        SGB(0x008,  4, 0);   // v + s WMMAs
    }

    const int node0 = tn * 16 + (hi ? 8 : 0);
    const float bbq0 = bQ[col0], bbq1 = bQ[col1];
    const float bbk0 = bK[col0], bbk1 = bK[col1];
    const float bbv0 = bV[col0], bbv1 = bV[col1];
    const float bbs0 = bS[col0], bbs1 = bS[col1];
    #pragma unroll
    for (int r = 0; r < 8; ++r){
        const size_t rowoff = (size_t)(node0 + r) * M;
        OQ[rowoff + col0] = aq0[r] + bbq0;  OQ[rowoff + col1] = aq1[r] + bbq1;
        OK[rowoff + col0] = ak0[r] + bbk0;  OK[rowoff + col1] = ak1[r] + bbk1;
        OV[rowoff + col0] = av0[r] + bbv0;  OV[rowoff + col1] = av1[r] + bbv1;
        OS[rowoff + col0] = as0[r] + bbs0;  OS[rowoff + col1] = as1[r] + bbs1;
    }
}

// ---------------- edge attention ----------------
// pass 1: alpha_e = dot(q[dst], k[src]) * scale ; segment max via atomicMax on encoded uint
__global__ void k_edge_dot(const float* __restrict__ q, const float* __restrict__ kk,
                           const int* __restrict__ src, const int* __restrict__ dst,
                           float* __restrict__ alpha, unsigned int* __restrict__ nmax,
                           int E, int C, float scale){
    int wid  = (blockIdx.x * blockDim.x + threadIdx.x) >> 5;
    int lane = threadIdx.x & 31;
    if (wid >= E) return;
    const int s = src[wid], d = dst[wid];
    float acc = 0.f;
    for (int c = lane; c < C; c += 32)
        acc += q[(size_t)d * C + c] * kk[(size_t)s * C + c];
    for (int off = 16; off; off >>= 1) acc += __shfl_xor(acc, off, 32);
    if (lane == 0){
        float a = acc * scale;
        alpha[wid] = a;
        atomicMax(nmax + d, enc_f(a));
    }
}
// pass 2: e = exp(alpha - max[dst]); segment sum
__global__ void k_edge_exp(float* __restrict__ alpha, const int* __restrict__ dst,
                           const unsigned int* __restrict__ nmax, float* __restrict__ nsum, int E){
    int e = blockIdx.x * blockDim.x + threadIdx.x;
    if (e >= E) return;
    const int d = dst[e];
    float ev = expf(alpha[e] - dec_f(nmax[d]));
    alpha[e] = ev;
    atomicAdd(nsum + d, ev);
}
// pass 3: out[dst] += (alpha/sum[dst]) * v[src]  (out pre-filled with skip projection)
__global__ void k_edge_agg(const float* __restrict__ alpha, const float* __restrict__ v,
                           const int* __restrict__ src, const int* __restrict__ dst,
                           const float* __restrict__ nsum, float* __restrict__ out,
                           int E, int C){
    int idx = blockIdx.x * blockDim.x + threadIdx.x;
    const int per = C >> 2;
    int e = idx / per;
    if (e >= E) return;
    int c = (idx - e * per) << 2;
    const int s = src[e], d = dst[e];
    const float coef = alpha[e] / nsum[d];
    const float4 vv = *(const float4*)(v + (size_t)s * C + c);
    float* o = out + (size_t)d * C + c;
    atomicAdd(o + 0, coef * vv.x);
    atomicAdd(o + 1, coef * vv.y);
    atomicAdd(o + 2, coef * vv.z);
    atomicAdd(o + 3, coef * vv.w);
}

// ---------------- global attention pooling ----------------
__global__ void k_gate(const float* __restrict__ x, const float* __restrict__ gw,
                       const float* __restrict__ gb, const int* __restrict__ batch,
                       float* __restrict__ gate, unsigned int* __restrict__ gmax, int N, int C){
    int n = blockIdx.x * blockDim.x + threadIdx.x;
    if (n >= N) return;
    float acc = gb[0];
    for (int c = 0; c < C; ++c) acc += x[(size_t)n * C + c] * gw[c];
    gate[n] = acc;
    atomicMax(gmax + batch[n], enc_f(acc));
}
__global__ void k_gate_exp(float* __restrict__ gate, const int* __restrict__ batch,
                           const unsigned int* __restrict__ gmax, float* __restrict__ gsum, int N){
    int n = blockIdx.x * blockDim.x + threadIdx.x;
    if (n >= N) return;
    const int b = batch[n];
    float ev = expf(gate[n] - dec_f(gmax[b]));
    gate[n] = ev;
    atomicAdd(gsum + b, ev);
}
__global__ void k_pool(const float* __restrict__ x, const float* __restrict__ gate,
                       const float* __restrict__ gsum, const int* __restrict__ batch,
                       float* __restrict__ pooled, int N, int C){
    int idx = blockIdx.x * blockDim.x + threadIdx.x;
    if (idx >= N * C) return;
    int n = idx / C, c = idx - n * C;
    const int b = batch[n];
    atomicAdd(pooled + (size_t)b * C + c, (gate[n] / gsum[b]) * x[(size_t)n * C + c]);
}
__global__ void k_fc(const float* __restrict__ pooled, const float* __restrict__ W,
                     const float* __restrict__ b, float* __restrict__ out, int G, int C, int O){
    int idx = blockIdx.x * blockDim.x + threadIdx.x;
    if (idx >= G * O) return;
    int g = idx / O, o = idx - g * O;
    float acc = b[o];
    for (int c = 0; c < C; ++c) acc += pooled[(size_t)g * C + c] * W[c * O + o];
    out[idx] = acc;
}

// ---------------- host orchestration ----------------
extern "C" void kernel_launch(void* const* d_in, const int* in_sizes, int n_in,
                              void* d_out, int out_size, void* d_ws, size_t ws_size,
                              hipStream_t stream) {
    (void)ws_size;
    const int dims[5] = {128, 512, 256, 64, 32};
    const int N = in_sizes[0] / 128;
    const int E = in_sizes[1] / 2;
    const int G = out_size / 2;
    const int MAXC = 512;

    const float* x_in  = (const float*)d_in[0];
    const int*   eidx  = (const int*)d_in[1];
    const int*   src   = eidx;
    const int*   dst   = eidx + E;
    const int*   batch = (const int*)d_in[2];

    const float *Wq[4], *bq[4], *Wk[4], *bk[4], *Wv[4], *bv[4], *Ws[4], *bs[4];
    const float *gate_W, *gate_b, *fc_W, *fc_b;
    if (n_in >= 39) {
        // each list element is its own input
        for (int l = 0; l < 4; ++l){
            Wq[l] = (const float*)d_in[3  + l];  bq[l] = (const float*)d_in[7  + l];
            Wk[l] = (const float*)d_in[11 + l];  bk[l] = (const float*)d_in[15 + l];
            Wv[l] = (const float*)d_in[19 + l];  bv[l] = (const float*)d_in[23 + l];
            Ws[l] = (const float*)d_in[27 + l];  bs[l] = (const float*)d_in[31 + l];
        }
        gate_W = (const float*)d_in[35]; gate_b = (const float*)d_in[36];
        fc_W   = (const float*)d_in[37]; fc_b   = (const float*)d_in[38];
    } else {
        // lists concatenated flat per dict entry (15 inputs)
        size_t wof[4], bof[4], wa = 0, ba = 0;
        for (int l = 0; l < 4; ++l){ wof[l] = wa; bof[l] = ba;
            wa += (size_t)dims[l] * dims[l + 1]; ba += dims[l + 1]; }
        for (int l = 0; l < 4; ++l){
            Wq[l] = (const float*)d_in[3] + wof[l];  bq[l] = (const float*)d_in[4] + bof[l];
            Wk[l] = (const float*)d_in[5] + wof[l];  bk[l] = (const float*)d_in[6] + bof[l];
            Wv[l] = (const float*)d_in[7] + wof[l];  bv[l] = (const float*)d_in[8] + bof[l];
            Ws[l] = (const float*)d_in[9] + wof[l];  bs[l] = (const float*)d_in[10] + bof[l];
        }
        gate_W = (const float*)d_in[11]; gate_b = (const float*)d_in[12];
        fc_W   = (const float*)d_in[13]; fc_b   = (const float*)d_in[14];
    }

    // workspace carve-out (256B aligned bump allocator)
    char* p = (char*)d_ws;
    auto alloc = [&](size_t bytes) -> char* {
        char* r = p; p += (bytes + 255) & ~(size_t)255; return r;
    };
    _Float16* xb   = (_Float16*)alloc((size_t)N * MAXC * sizeof(_Float16));
    float* qbuf    = (float*)alloc((size_t)N * MAXC * sizeof(float));
    float* kbuf    = (float*)alloc((size_t)N * MAXC * sizeof(float));
    float* vbuf    = (float*)alloc((size_t)N * MAXC * sizeof(float));
    float* xpingA  = (float*)alloc((size_t)N * MAXC * sizeof(float));
    float* xpingB  = (float*)alloc((size_t)N * MAXC * sizeof(float));
    float* alpha   = (float*)alloc((size_t)E * sizeof(float));
    unsigned int* nmax = (unsigned int*)alloc((size_t)N * sizeof(unsigned int));
    float* nsum    = (float*)alloc((size_t)N * sizeof(float));
    float* gate    = (float*)alloc((size_t)N * sizeof(float));
    _Float16* wtq  = (_Float16*)alloc(131072 * sizeof(_Float16));
    _Float16* wtk  = (_Float16*)alloc(131072 * sizeof(_Float16));
    _Float16* wtv  = (_Float16*)alloc(131072 * sizeof(_Float16));
    _Float16* wts  = (_Float16*)alloc(131072 * sizeof(_Float16));
    float* pooled  = (float*)alloc((size_t)G * 32 * sizeof(float));
    unsigned int* gmax = (unsigned int*)alloc((size_t)G * sizeof(unsigned int));
    float* gsum    = (float*)alloc((size_t)G * sizeof(float));

    const unsigned int ENC_NEG_INF = 0x007FFFFFu;   // enc_f(-inf)
    const int TB = 256;

    float* xcur = xpingA;
    float* xnxt = xpingB;

    for (int l = 0; l < 4; ++l){
        const int din = dims[l], dout = dims[l + 1];
        const float* layer_in = (l == 0) ? x_in : xcur;

        // activations -> f16
        k_f32_to_f16<<<cdiv((long long)N * din, TB), TB, 0, stream>>>(layer_in, xb, N * din);
        // weights -> transposed f16 (M x K)
        int wn = din * dout;
        k_transpose_f16<<<cdiv(wn, TB), TB, 0, stream>>>(Wq[l], wtq, din, dout);
        k_transpose_f16<<<cdiv(wn, TB), TB, 0, stream>>>(Wk[l], wtk, din, dout);
        k_transpose_f16<<<cdiv(wn, TB), TB, 0, stream>>>(Wv[l], wtv, din, dout);
        k_transpose_f16<<<cdiv(wn, TB), TB, 0, stream>>>(Ws[l], wts, din, dout);

        // fused dense projections via WMMA (q, k, v, skip in one pass over A)
        long long waves = (long long)((N + 15) / 16) * (dout / 32);
        int gblocks = cdiv(waves * 32, TB);
        k_gemm_qkvs<<<gblocks, TB, 0, stream>>>(xb, wtq, wtk, wtv, wts,
                                                bq[l], bk[l], bv[l], bs[l],
                                                qbuf, kbuf, vbuf, xnxt,
                                                N, din, dout);

        // segment softmax over incoming edges
        k_fill_u32<<<cdiv(N, TB), TB, 0, stream>>>(nmax, ENC_NEG_INF, N);
        k_fill_f32<<<cdiv(N, TB), TB, 0, stream>>>(nsum, 0.f, N);
        float scale = 1.f / sqrtf((float)dout);
        k_edge_dot<<<cdiv((long long)E * 32, TB), TB, 0, stream>>>(qbuf, kbuf, src, dst,
                                                                   alpha, nmax, E, dout, scale);
        k_edge_exp<<<cdiv(E, TB), TB, 0, stream>>>(alpha, dst, nmax, nsum, E);
        k_edge_agg<<<cdiv((long long)E * (dout / 4), TB), TB, 0, stream>>>(alpha, vbuf, src, dst,
                                                                           nsum, xnxt, E, dout);
        // elu(agg + skip)
        k_elu<<<cdiv((long long)N * dout, TB), TB, 0, stream>>>(xnxt, N * dout);

        float* t = xcur; xcur = xnxt; xnxt = t;
    }

    // global attention pooling over batch (xcur is N x 32)
    k_fill_u32<<<1, 64, 0, stream>>>(gmax, ENC_NEG_INF, G);
    k_fill_f32<<<1, 64, 0, stream>>>(gsum, 0.f, G);
    k_fill_f32<<<cdiv(G * 32, TB), TB, 0, stream>>>(pooled, 0.f, G * 32);
    k_gate<<<cdiv(N, TB), TB, 0, stream>>>(xcur, gate_W, gate_b, batch, gate, gmax, N, 32);
    k_gate_exp<<<cdiv(N, TB), TB, 0, stream>>>(gate, batch, gmax, gsum, N);
    k_pool<<<cdiv((long long)N * 32, TB), TB, 0, stream>>>(xcur, gate, gsum, batch, pooled, N, 32);
    k_fc<<<1, 128, 0, stream>>>(pooled, fc_W, fc_b, (float*)d_out, G, 32, 2);
}
